// SymmetricClipLoss_91018946937348
// MI455X (gfx1250) — matrix operand order repaired and checked by reading, not compiled, not run on hardware
//
#include <hip/hip_runtime.h>

// Symmetric CLIP loss, fused: never materializes the 16384x16384 logits.
//   pass1: rowstats(ref_n, cand_n)  -> sumExpRow, argmaxRow
//   pass2: rowstats(cand_n, ref_n)  -> sumExpCol
//   loss  = 0.5*mean(1/T + log(sumExpRow) - d) + 0.5*mean(1/T + log(sumExpCol) - d)
// GEMM core: v_wmma_f32_16x16x32_bf16, 32(M)x64(N) tile per wave so each
// B fragment feeds two WMMAs (12 b128 loads per 8 WMMAs instead of 10 per 4).

typedef unsigned short u16;
typedef unsigned int   u32;
typedef unsigned long long u64;

typedef __attribute__((ext_vector_type(16))) __bf16 v16bf;
typedef __attribute__((ext_vector_type(8)))  float  v8f;

#define N_ROWS 16384
#define DIMS   512
#define INV_T  14.285714285714286f   /* 1/0.07 ; also max possible logit */

union BFrag { v16bf v; uint4 u[2]; };

__device__ __forceinline__ u16 f2bf(float f) {
  u32 u = __float_as_uint(f);
  u32 r = (u + 0x7FFFu + ((u >> 16) & 1u)) >> 16;   // round-to-nearest-even
  return (u16)r;
}

__device__ __forceinline__ u32 orderable(float f) {
  u32 u = __float_as_uint(f);
  return (u & 0x80000000u) ? ~u : (u | 0x80000000u);
}

// ---------------------------------------------------------------------------
// 1) L2-normalize both matrices to bf16; compute fp32 diagonal logit d[i].
//    One wave per row (wave32), 16 elements per lane.
// ---------------------------------------------------------------------------
__global__ __launch_bounds__(256) void clip_normalize(
    const float* __restrict__ ref, const float* __restrict__ cand,
    u16* __restrict__ refn, u16* __restrict__ candn, float* __restrict__ dvec)
{
  const int lane = threadIdx.x & 31;
  const int wave = threadIdx.x >> 5;
  const int row  = blockIdx.x * 8 + wave;
  const float* rp = ref  + (size_t)row * DIMS;
  const float* cp = cand + (size_t)row * DIMS;

  float r[16], c[16], sr = 0.f, sc = 0.f;
#pragma unroll
  for (int e = 0; e < 16; ++e) {
    r[e] = rp[lane + 32 * e]; sr += r[e] * r[e];
    c[e] = cp[lane + 32 * e]; sc += c[e] * c[e];
  }
  for (int mask = 16; mask >= 1; mask >>= 1) {
    sr += __shfl_xor(sr, mask, 32);
    sc += __shfl_xor(sc, mask, 32);
  }
  const float ir = 1.f / fmaxf(sqrtf(sr), 1e-12f);
  const float ic = 1.f / fmaxf(sqrtf(sc), 1e-12f);

  float dot = 0.f;
#pragma unroll
  for (int e = 0; e < 16; ++e) {
    float rn = r[e] * ir, cn = c[e] * ic;
    refn [(size_t)row * DIMS + lane + 32 * e] = f2bf(rn);
    candn[(size_t)row * DIMS + lane + 32 * e] = f2bf(cn);
    dot += rn * cn;
  }
  for (int mask = 16; mask >= 1; mask >>= 1) dot += __shfl_xor(dot, mask, 32);
  if (lane == 0) dvec[row] = dot * INV_T;
}

// ---------------------------------------------------------------------------
// 2) Zero the reduction buffers (d_ws is poisoned).
// ---------------------------------------------------------------------------
__global__ void clip_init(float* sumR, float* sumC, u64* keys) {
  int i = blockIdx.x * 256 + threadIdx.x;
  if (i < N_ROWS) { sumR[i] = 0.f; sumC[i] = 0.f; keys[i] = 0ull; }
}

// ---------------------------------------------------------------------------
// 3) Fused GEMM + softmax-stat kernel. 32x64 tile per wave.
//    Each wave owns 32 rows of A (two A fragments), scans colsPerSplit
//    columns of B in 64-wide tiles. Per K-step: 2 A frags + 4 B frags,
//    8 WMMAs (each B frag reused by both A frags). Epilogue: fixed-max
//    softmax accumulation + running argmax; cross-split combine via atomics.
//
//    Fragment layouts (CDNA5 ISA 7.12.2, wave32):
//      A 16x32 bf16: lane l holds row m=l&15; elems 0-7 = K(k0+8*hi .. +7),
//                    elems 8-15 = K(k0+8*hi+16 .. +23), hi=l>>4.
//      B 32x16 bf16: lane l holds col n=l&15; same K split as A.
//      C/D 16x16 f32: VGPR r, lane l -> row r+8*hi, col l&15.
// ---------------------------------------------------------------------------
__global__ __launch_bounds__(256) void clip_rowstats(
    const u16* __restrict__ A, const u16* __restrict__ B,
    float* __restrict__ sumExp, u64* __restrict__ keys, int colsPerSplit)
{
  const int lane = threadIdx.x & 31;
  const int wave = threadIdx.x >> 5;
  const int hi   = lane >> 4;
  const int m    = lane & 15;
  const int i0   = blockIdx.x * 256 + wave * 32;   // 32 rows per wave
  const int cBeg = blockIdx.y * colsPerSplit;

  const u16* aRow0 = A + (size_t)(i0 + m)      * DIMS + 8 * hi;
  const u16* aRow1 = A + (size_t)(i0 + 16 + m) * DIMS + 8 * hi;

  // Stats for 16 (row-tile u, vgpr r) pairs: entry e = u*8 + r
  float partial[16]; float mv[16]; int mi[16];
#pragma unroll
  for (int e = 0; e < 16; ++e) { partial[e] = 0.f; mv[e] = -3.0e38f; mi[e] = 0; }

  for (int c0 = cBeg; c0 < cBeg + colsPerSplit; c0 += 64) {
    v8f acc0[4] = {};   // rows i0    .. i0+15
    v8f acc1[4] = {};   // rows i0+16 .. i0+31
    const u16* bCol = B + (size_t)(c0 + m) * DIMS + 8 * hi;

#pragma unroll 4
    for (int ks = 0; ks < 16; ++ks) {
      const int k0 = ks * 32;
      BFrag a0, a1;
      a0.u[0] = *(const uint4*)(aRow0 + k0);
      a0.u[1] = *(const uint4*)(aRow0 + k0 + 16);
      a1.u[0] = *(const uint4*)(aRow1 + k0);
      a1.u[1] = *(const uint4*)(aRow1 + k0 + 16);
#pragma unroll
      for (int t = 0; t < 4; ++t) {
        BFrag b;
        const uint4* p = (const uint4*)(bCol + (size_t)(16 * t) * DIMS + k0);
        b.u[0] = p[0];
        b.u[1] = p[2];
        acc0[t] = __builtin_amdgcn_wmma_f32_16x16x32_bf16(
            false, a0.v, false, b.v, (short)0, acc0[t], false, false);
        acc1[t] = __builtin_amdgcn_wmma_f32_16x16x32_bf16(
            false, a1.v, false, b.v, (short)0, acc1[t], false, false);
      }
    }

    // Epilogue: max logit == 1/T (|dot| <= 1 for unit vectors), so no online
    // max rescaling is needed: accumulate exp(logit - 1/T) directly.
#pragma unroll
    for (int t = 0; t < 4; ++t) {
      const int col = c0 + 16 * t + m;
#pragma unroll
      for (int r = 0; r < 8; ++r) {
        float v0 = acc0[t][r];
        partial[r] += __expf(v0 * INV_T - INV_T);
        if (v0 > mv[r]) { mv[r] = v0; mi[r] = col; }
        float v1 = acc1[t][r];
        partial[8 + r] += __expf(v1 * INV_T - INV_T);
        if (v1 > mv[8 + r]) { mv[8 + r] = v1; mi[8 + r] = col; }
      }
    }
  }

  // Cross-lane reduce within each 16-lane half (entry e -> row
  // i0 + 16*(e>>3) + (e&7) + 8*hi), then atomics across column splits.
#pragma unroll
  for (int e = 0; e < 16; ++e) {
    float s = partial[e]; float v = mv[e]; int i = mi[e];
    for (int mask = 8; mask >= 1; mask >>= 1) {
      s += __shfl_xor(s, mask, 32);
      float ov = __shfl_xor(v, mask, 32);
      int   oi = __shfl_xor(i, mask, 32);
      if (ov > v || (ov == v && oi < i)) { v = ov; i = oi; }
    }
    if (m == 0) {
      const int row = i0 + 16 * (e >> 3) + (e & 7) + 8 * hi;
      atomicAdd(&sumExp[row], s);
      if (keys) {
        // larger value wins; on ties the ~idx low bits prefer the smaller idx
        u64 key = ((u64)orderable(v) << 32) | (u32)(~(u32)i);
        atomicMax(&keys[row], key);
      }
    }
  }
}

// ---------------------------------------------------------------------------
// 4) Final reduction: loss + corrects -> d_out[0..1].
// ---------------------------------------------------------------------------
__global__ __launch_bounds__(256) void clip_finalize(
    const float* __restrict__ sumR, const float* __restrict__ sumC,
    const float* __restrict__ dvec, const u64* __restrict__ keys,
    float* __restrict__ out)
{
  __shared__ float s1[256];
  __shared__ float s2[256];
  __shared__ int   sc_[256];
  float l1 = 0.f, l2 = 0.f; int corr = 0;
  for (int i = threadIdx.x; i < N_ROWS; i += 256) {
    float d = dvec[i];
    l1 += (INV_T + logf(sumR[i])) - d;
    l2 += (INV_T + logf(sumC[i])) - d;
    u32 idx = ~(u32)(keys[i] & 0xFFFFFFFFull);
    corr += (idx == (u32)i) ? 1 : 0;
  }
  s1[threadIdx.x] = l1; s2[threadIdx.x] = l2; sc_[threadIdx.x] = corr;
  __syncthreads();
  for (int off = 128; off > 0; off >>= 1) {
    if (threadIdx.x < off) {
      s1[threadIdx.x]  += s1[threadIdx.x + off];
      s2[threadIdx.x]  += s2[threadIdx.x + off];
      sc_[threadIdx.x] += sc_[threadIdx.x + off];
    }
    __syncthreads();
  }
  if (threadIdx.x == 0) {
    out[0] = 0.5f * (s1[0] + s2[0]) / (float)N_ROWS;
    out[1] = (float)sc_[0];
  }
}

// ---------------------------------------------------------------------------
extern "C" void kernel_launch(void* const* d_in, const int* in_sizes, int n_in,
                              void* d_out, int out_size, void* d_ws, size_t ws_size,
                              hipStream_t stream) {
  (void)in_sizes; (void)n_in; (void)out_size; (void)ws_size;
  const float* ref  = (const float*)d_in[0];
  const float* cand = (const float*)d_in[1];

  char* ws = (char*)d_ws;
  const size_t bfBytes = (size_t)N_ROWS * DIMS * sizeof(u16);   // 16 MiB each
  u16*   refn  = (u16*)ws;
  u16*   candn = (u16*)(ws + bfBytes);
  float* dvec  = (float*)(ws + 2 * bfBytes);
  float* sumR  = dvec + N_ROWS;
  float* sumC  = sumR + N_ROWS;
  u64*   keys  = (u64*)(sumC + N_ROWS);   // 32MiB + 192KiB offset: 8B aligned

  clip_normalize<<<N_ROWS / 8, 256, 0, stream>>>(ref, cand, refn, candn, dvec);
  clip_init<<<(N_ROWS + 255) / 256, 256, 0, stream>>>(sumR, sumC, keys);

  const int splits = 8;                 // 64 row-blocks x 8 col-splits = 512 WGs
  clip_rowstats<<<dim3(N_ROWS / 256, splits), 256, 0, stream>>>(
      refn, candn, sumR, keys, N_ROWS / splits);
  clip_rowstats<<<dim3(N_ROWS / 256, splits), 256, 0, stream>>>(
      candn, refn, sumC, nullptr, N_ROWS / splits);

  clip_finalize<<<1, 256, 0, stream>>>(sumR, sumC, dvec, keys, (float*)d_out);
}
